// PoolHiddenNet_d_14568529068508
// MI455X (gfx1250) — matrix operand order repaired and checked by reading, not compiled, not run on hardware
//
#include <hip/hip_runtime.h>
#include <hip/hip_bf16.h>

// ---------------------------------------------------------------------------
// PoolHiddenNet fused pipeline for MI455X (gfx1250, wave32, WMMA bf16).
//
// Compute:  ~77 GFLOP dense GEMM vs <200MB traffic  -> compute bound
// Precision: bf16 A/B with f32 accumulate (V_WMMA_F32_16X16X32_BF16)
// b1/b2 cancel inside training-mode BatchNorm (constant per column) -> dropped
// A-tiles staged via GLOBAL_LOAD_ASYNC_TO_LDS_B128 (ASYNCcnt) when the
// toolchain exposes the builtin; falls back to global_load+ds_store otherwise.
// ---------------------------------------------------------------------------

typedef __bf16 bf16;
typedef __attribute__((ext_vector_type(16))) __bf16 v16bf;
typedef __attribute__((ext_vector_type(8)))  __bf16 v8bf;
typedef __attribute__((ext_vector_type(8)))  float  v8f;
typedef __attribute__((ext_vector_type(4)))  int    v4i;
// builtin takes v4i pointers in explicit address spaces (AS1 global, AS3 LDS)
typedef __attribute__((address_space(1))) v4i as1_v4i;
typedef __attribute__((address_space(3))) v4i as3_v4i;

#if __has_builtin(__builtin_amdgcn_global_load_async_to_lds_b128) && \
    __has_builtin(__builtin_amdgcn_s_wait_asynccnt)
#define USE_ASYNC_LDS 1
#else
#define USE_ASYNC_LDS 0
#endif

#define SCENES 64
#define NPED   32
#define ROWS   65536          // SCENES * NPED * NPED
#define KX     128            // E + H
#define DD1    512
#define DD2    1024
#define BNEPS  1e-5f

// workspace layout (bytes)
#define OFF_W1T   0u          //  512*128 bf16  =    131072 B
#define OFF_W2T   131072u     // 1024*512 bf16  =   1048576 B
#define OFF_X     1179648u    // 65536*128 bf16 =  16777216 B
#define OFF_A1    17956864u   // 65536*512 bf16 =  67108864 B
#define OFF_STATS 85065728u   // 2*65536 f32    =    524288 B
// total ~ 85.6 MB

// ---------------------------------------------------------------------------
// Prep: bf16-transpose weights (Wt[n][k] = W[k][n]) + zero BN2 stats
// ---------------------------------------------------------------------------
__global__ void prep_misc(const float* __restrict__ W1, const float* __restrict__ W2,
                          bf16* __restrict__ W1t, bf16* __restrict__ W2t,
                          float* __restrict__ stats) {
    int idx = blockIdx.x * blockDim.x + threadIdx.x;
    if (idx < 65536) {                       // W1: (128 x 512) -> W1t (512 x 128)
        int n = idx >> 7, k = idx & 127;
        W1t[idx] = (bf16)W1[k * DD1 + n];
    } else if (idx < 65536 + 524288) {       // W2: (512 x 1024) -> W2t (1024 x 512)
        int i = idx - 65536;
        int n = i >> 9, k = i & 511;
        W2t[i] = (bf16)W2[k * DD2 + n];
    } else if (idx < 65536 + 524288 + 131072) {
        stats[idx - 589824] = 0.f;
    }
}

// ---------------------------------------------------------------------------
// Prep: X[row=(s,j,k)][0:64]=emb(pos_k-pos_j), [64:128]=h_k   (bf16)
// ---------------------------------------------------------------------------
__global__ void prep_x(const float* __restrict__ h, const float* __restrict__ pos,
                       const float* __restrict__ Wemb, const float* __restrict__ bemb,
                       bf16* __restrict__ X) {
    int idx = blockIdx.x * blockDim.x + threadIdx.x;   // ROWS * 64 (2 cols each)
    if (idx >= ROWS * 64) return;
    int R  = idx >> 6, cp = idx & 63;
    int s  = R >> 10, rr = R & 1023;
    int j  = rr >> 5, k = rr & 31;
    int c0 = cp * 2;
    float o0, o1;
    if (c0 < 64) {
        float rx = pos[(s * 32 + k) * 2]     - pos[(s * 32 + j) * 2];
        float ry = pos[(s * 32 + k) * 2 + 1] - pos[(s * 32 + j) * 2 + 1];
        o0 = rx * Wemb[c0]     + ry * Wemb[64 + c0]     + bemb[c0];
        o1 = rx * Wemb[c0 + 1] + ry * Wemb[64 + c0 + 1] + bemb[c0 + 1];
    } else {
        int c = c0 - 64;
        o0 = h[(s * 32 + k) * 64 + c];
        o1 = h[(s * 32 + k) * 64 + c + 1];
    }
    X[(size_t)R * KX + c0]     = (bf16)o0;
    X[(size_t)R * KX + c0 + 1] = (bf16)o1;
}

// ---------------------------------------------------------------------------
// BN1 (+ReLU) in place: one thread per (scene, col); stats over 1024 rows.
// ---------------------------------------------------------------------------
__global__ void bn1_kernel(bf16* __restrict__ A1, const float* __restrict__ g1,
                           const float* __restrict__ beta1) {
    int idx = blockIdx.x * blockDim.x + threadIdx.x;
    if (idx >= SCENES * DD1) return;
    int s = idx / DD1, c = idx % DD1;
    const size_t base = (size_t)s * 1024 * DD1 + c;
    float s1 = 0.f, s2 = 0.f;
    for (int r = 0; r < 1024; ++r) {
        float v = (float)A1[base + (size_t)r * DD1];
        s1 += v; s2 += v * v;
    }
    float mean = s1 * (1.f / 1024.f);
    float var  = s2 * (1.f / 1024.f) - mean * mean;   // biased, matches reference
    float inv  = rsqrtf(var + BNEPS);
    float sc = g1[c] * inv;
    float sh = beta1[c] - mean * sc;
    for (int r = 0; r < 1024; ++r) {
        size_t o = base + (size_t)r * DD1;
        float v = (float)A1[o] * sc + sh;
        A1[o] = (bf16)fmaxf(v, 0.f);
    }
}

// ---------------------------------------------------------------------------
// Fold BN2 stats into per-(scene,col) scale/shift (overwrite stats buffer)
// ---------------------------------------------------------------------------
__global__ void bn2fin(float* __restrict__ stats, const float* __restrict__ g2,
                       const float* __restrict__ beta2) {
    int idx = blockIdx.x * blockDim.x + threadIdx.x;
    if (idx >= SCENES * DD2) return;
    int c = idx & 1023;
    float mean = stats[idx] * (1.f / 1024.f);
    float var  = stats[65536 + idx] * (1.f / 1024.f) - mean * mean;
    float inv  = rsqrtf(var + BNEPS);
    float sc = g2[c] * inv;
    stats[idx]         = sc;
    stats[65536 + idx] = beta2[c] - mean * sc;
}

// ---------------------------------------------------------------------------
// bf16 WMMA GEMM: C(128x128 per WG) = A(rows x KDIM) * Bt(n x KDIM)^T
// 4 waves (2x2), each wave 64x64 via 4x4 tiles of v_wmma_f32_16x16x32_bf16.
// MODE 0: store bf16 (GEMM1 -> H1)
// MODE 1: epilogue accumulates per-scene column sum/sum^2 (atomics)
// MODE 2: epilogue fuses BN2 scale/shift + ReLU + max-over-k -> d_out
// ---------------------------------------------------------------------------
template<int KDIM, int NDIM, int MODE>
__global__ __launch_bounds__(128)
void gemm_bf16(const bf16* __restrict__ A, const bf16* __restrict__ Bt,
               bf16* __restrict__ Obf, float* __restrict__ stats,
               float* __restrict__ Ofin) {
    // padded stride 40 bf16 (80B, 16B-aligned rows, conflict-free b128 reads)
    __shared__ __attribute__((aligned(32))) bf16 lA[128 * 40];

    const int tid  = threadIdx.x;
    const int lane = tid & 31, wave = tid >> 5;
    const int wr = wave >> 1, wc = wave & 1;
    const int h  = lane >> 4, m = lane & 15;
    const int rowbase = blockIdx.x * 128;
    const int colbase = blockIdx.y * 128;

    v8f acc[4][4];
    const v8f vzero = {0.f, 0.f, 0.f, 0.f, 0.f, 0.f, 0.f, 0.f};
    #pragma unroll
    for (int t = 0; t < 4; ++t)
        #pragma unroll
        for (int c = 0; c < 4; ++c) acc[t][c] = vzero;

    for (int kc = 0; kc < KDIM; kc += 32) {
        __syncthreads();
        // cooperative A tile load: 128 rows x 32 k (8KB) as 512 x 16B chunks
        #pragma unroll
        for (int q = 0; q < 4; ++q) {
            int idx = tid + q * 128;
            int row = idx >> 2, c16 = idx & 3;
            const bf16* gsrc = A + (size_t)(rowbase + row) * KDIM + kc + c16 * 8;
            bf16* ldst = lA + row * 40 + c16 * 8;
#if USE_ASYNC_LDS
            // direct memory -> LDS DMA, tracked by ASYNCcnt (no VGPR staging)
            __builtin_amdgcn_global_load_async_to_lds_b128(
                (as1_v4i*)gsrc, (as3_v4i*)ldst, 0, 0);
#else
            *(v8bf*)ldst = *(const v8bf*)gsrc;
#endif
        }
#if USE_ASYNC_LDS
        __builtin_amdgcn_s_wait_asynccnt(0);
#endif
        __syncthreads();

        // B fragments straight from pre-transposed bf16 weights (L2-resident):
        // lane n holds K chunks kc+8h and kc+16+8h (2 x b128)
        v16bf bfrag[4];
        #pragma unroll
        for (int ct = 0; ct < 4; ++ct) {
            int nn = colbase + wc * 64 + ct * 16 + m;
            union { v16bf v; v8bf hv[2]; } b;
            b.hv[0] = *(const v8bf*)(Bt + (size_t)nn * KDIM + kc + 8 * h);
            b.hv[1] = *(const v8bf*)(Bt + (size_t)nn * KDIM + kc + 16 + 8 * h);
            bfrag[ct] = b.v;
        }

        #pragma unroll
        for (int t = 0; t < 4; ++t) {
            // A fragment: lane m = row; K chunks at 8h and 16+8h (2 x b128)
            const bf16* ar = lA + (wr * 64 + t * 16 + m) * 40;
            union { v16bf v; v8bf hv[2]; } a;
            a.hv[0] = *(const v8bf*)(ar + 8 * h);
            a.hv[1] = *(const v8bf*)(ar + 16 + 8 * h);
            #pragma unroll
            for (int ct = 0; ct < 4; ++ct) {
                acc[t][ct] = __builtin_amdgcn_wmma_f32_16x16x32_bf16(
                    false, a.v, false, bfrag[ct], (short)0, acc[t][ct],
                    false, false);
            }
        }
    }

    // C/D layout: VGPR r holds M = r + 8h ; N = lane & 15
    if (MODE == 0) {
        #pragma unroll
        for (int t = 0; t < 4; ++t)
            #pragma unroll
            for (int ct = 0; ct < 4; ++ct) {
                int col = colbase + wc * 64 + ct * 16 + m;
                #pragma unroll
                for (int r = 0; r < 8; ++r) {
                    int row = rowbase + wr * 64 + t * 16 + r + 8 * h;
                    Obf[(size_t)row * NDIM + col] = (bf16)acc[t][ct][r];
                }
            }
    } else if (MODE == 1) {
        int scene = rowbase >> 10;
        #pragma unroll
        for (int ct = 0; ct < 4; ++ct) {
            float s1 = 0.f, s2 = 0.f;
            #pragma unroll
            for (int t = 0; t < 4; ++t)
                #pragma unroll
                for (int r = 0; r < 8; ++r) {
                    float v = acc[t][ct][r];
                    s1 += v; s2 += v * v;
                }
            s1 += __shfl_xor(s1, 16);
            s2 += __shfl_xor(s2, 16);
            if (lane < 16) {
                int col = colbase + wc * 64 + ct * 16 + lane;
                atomicAdd(stats + scene * DD2 + col, s1);
                atomicAdd(stats + 65536 + scene * DD2 + col, s2);
            }
        }
    } else {  // MODE == 2: BN2 + ReLU + max over k (128 rows = 4 full k-groups)
        int scene = rowbase >> 10;
        #pragma unroll
        for (int ct = 0; ct < 4; ++ct) {
            int col = colbase + wc * 64 + ct * 16 + m;
            float sc = stats[scene * DD2 + col];
            float sh = stats[65536 + scene * DD2 + col];
            #pragma unroll
            for (int jg = 0; jg < 2; ++jg) {
                float mx = 0.f;  // ReLU output >= 0, so 0 is a safe identity
                #pragma unroll
                for (int tt = 0; tt < 2; ++tt) {
                    int t = jg * 2 + tt;
                    #pragma unroll
                    for (int r = 0; r < 8; ++r) {
                        float v = fmaxf(acc[t][ct][r] * sc + sh, 0.f);
                        mx = fmaxf(mx, v);
                    }
                }
                mx = fmaxf(mx, __shfl_xor(mx, 16));
                if (lane < 16) {
                    int j = ((rowbase & 1023) >> 5) + wr * 2 + jg;
                    Ofin[(size_t)(scene * NPED + j) * DD2 + colbase + wc * 64 +
                         ct * 16 + lane] = mx;
                }
            }
        }
    }
}

// ---------------------------------------------------------------------------
extern "C" void kernel_launch(void* const* d_in, const int* in_sizes, int n_in,
                              void* d_out, int out_size, void* d_ws, size_t ws_size,
                              hipStream_t stream) {
    (void)in_sizes; (void)n_in; (void)out_size; (void)ws_size;
    const float* h_states = (const float*)d_in[0];
    const float* end_pos  = (const float*)d_in[2];
    const float* W_emb    = (const float*)d_in[6];
    const float* b_emb    = (const float*)d_in[7];
    const float* W1       = (const float*)d_in[8];
    // b1 (d_in[9]) cancels inside BN1 -> unused
    const float* g1       = (const float*)d_in[10];
    const float* beta1    = (const float*)d_in[11];
    const float* W2       = (const float*)d_in[12];
    // b2 (d_in[13]) cancels inside BN2 -> unused
    const float* g2       = (const float*)d_in[14];
    const float* beta2    = (const float*)d_in[15];

    char*  ws    = (char*)d_ws;
    bf16*  W1t   = (bf16*)(ws + OFF_W1T);
    bf16*  W2t   = (bf16*)(ws + OFF_W2T);
    bf16*  X     = (bf16*)(ws + OFF_X);
    bf16*  A1    = (bf16*)(ws + OFF_A1);
    float* stats = (float*)(ws + OFF_STATS);
    float* out   = (float*)d_out;

    prep_misc<<<(720896 + 255) / 256, 256, 0, stream>>>(W1, W2, W1t, W2t, stats);
    prep_x<<<(ROWS * 64 + 255) / 256, 256, 0, stream>>>(h_states, end_pos, W_emb,
                                                        b_emb, X);
    // GEMM1: (65536 x 128) * (128 x 512) -> H1 bf16 (staged in A1 buffer)
    gemm_bf16<KX, DD1, 0><<<dim3(512, 4), 128, 0, stream>>>(X, W1t, A1, stats,
                                                            nullptr);
    bn1_kernel<<<(SCENES * DD1 + 255) / 256, 256, 0, stream>>>(A1, g1, beta1);
    // GEMM2 pass A: stats only (per-scene column sums via atomics)
    gemm_bf16<DD1, DD2, 1><<<dim3(512, 8), 128, 0, stream>>>(A1, W2t, nullptr,
                                                             stats, nullptr);
    bn2fin<<<(SCENES * DD2 + 255) / 256, 256, 0, stream>>>(stats, g2, beta2);
    // GEMM2 pass B: recompute + fused BN2/ReLU/max-over-k -> d_out
    gemm_bf16<DD1, DD2, 2><<<dim3(512, 8), 128, 0, stream>>>(A1, W2t, nullptr,
                                                             stats, out);
}
